// ESN_652835029763
// MI455X (gfx1250) — matrix operand order repaired and checked by reading, not compiled, not run on hardware
//
#include <hip/hip_runtime.h>
#include <hip/hip_bf16.h>

// ---------------------------------------------------------------------------
// ESN reservoir scan for MI455X (gfx1250, wave32, WMMA).
//
// Strategy:
//  * Serial chain is ONLY the recurrence GEMM x@W^T (134 MFLOP/step),
//    done with v_wmma_f32_16x16x32_bf16, fp32 accumulation.
//  * x^T is logged per step (bf16) into workspace; X_XT and D_XT become
//    two post-scan GEMMs with K = (T - trans_len)*N_B = 30720.
//  * Row-major weight matrices are natively the K-major layout WMMA B-frags
//    want -> prep kernel only converts fp32->bf16, no transposes of weights.
//  * y_t computed one step late (wave 4, WGs 0..15) off the critical path;
//    final kernel emits y_{T-1} and x_n.
// ---------------------------------------------------------------------------

typedef __bf16 bf16;
typedef __attribute__((ext_vector_type(16))) __bf16 v16bf;
typedef __attribute__((ext_vector_type(8)))  float  v8f;

union FragAB { v16bf v; uint4 q[2]; };
union Pack8  { bf16 h[8]; uint4 q; };

#define N_U 64
#define N_X 1024
#define N_Y 64
#define T_STEPS 512
#define N_B 64
#define ALPHA 0.95f

// WMMA fragment loading recipe (wave32):
//  A (16x32, M x K): lane l: row m = l&15; VGPR j holds K-pair. Low half-wave
//    covers K {0..7,16..23}, high half {8..15,24..31}. From row-major [M][K]
//    storage this is two contiguous 16B loads at k0+(hi?8:0) and +16.
//  B (32x16, K x N): lane l: col n = l&15; low half-wave K {0..7,16..23}, high {8..15,24..31}.
//    From N-major [N][K] storage (== row-major weights W[n][k]) this is
//    two contiguous 16B loads at k0+(hi?8:0) and +16.

__device__ __forceinline__ v8f wmma_bf16(const FragAB& a, const FragAB& b, v8f c) {
  return __builtin_amdgcn_wmma_f32_16x16x32_bf16(false, a.v, false, b.v,
                                                 (short)0, c, false, false);
}

// ---------------------------------------------------------------------------
// Prep: fp32->bf16 weight conversion, zero initial state, build d^T history.
// ---------------------------------------------------------------------------
__global__ __launch_bounds__(256) void esn_prep(
    const float* __restrict__ W, const float* __restrict__ Win,
    const float* __restrict__ Wout, const float* __restrict__ target,
    bf16* __restrict__ Wbf, bf16* __restrict__ WinBf, bf16* __restrict__ WoutBf,
    bf16* __restrict__ xinit, bf16* __restrict__ dT)
{
  int gid = blockIdx.x * blockDim.x + threadIdx.x;
  int stride = gridDim.x * blockDim.x;
  for (int i = gid; i < N_X * N_X; i += stride) Wbf[i] = (bf16)W[i];
  for (int i = gid; i < N_X * N_U; i += stride) WinBf[i] = (bf16)Win[i];
  for (int i = gid; i < N_Y * N_X; i += stride) WoutBf[i] = (bf16)Wout[i];
  for (int i = gid; i < N_B * N_X; i += stride) xinit[i] = (bf16)0.0f;
  // dT[t][y][b] = target[t][b][y]  (A-fragment layout for D_XT GEMM)
  for (int i = gid; i < T_STEPS * N_B * N_Y; i += stride) {
    int t = i >> 12, r = i & 4095, y = r >> 6, b = r & 63;
    dT[i] = (bf16)target[(t << 12) + (b << 6) + y];
  }
}

// ---------------------------------------------------------------------------
// One recurrence step. Grid 64 x 256. Waves 0-3: GEMM1 (one 16-row M-tile
// each, N-tile = blockIdx.x, K = 1024 (W) + 64 (W_in)) + tanh/leaky epilogue,
// writes x row-major + x^T history slice. Wave 4 in WGs 0-15: y_{t-1}.
// ---------------------------------------------------------------------------
__global__ __launch_bounds__(256) void esn_step(
    const float* __restrict__ u,       // [N_B, N_U] step input
    const bf16*  __restrict__ xprev,   // [N_B, N_X]
    bf16*        __restrict__ xnew,    // [N_B, N_X]
    bf16*        __restrict__ xT_t,    // [N_X, N_B] history slice for step t
    const bf16*  __restrict__ Wbf,     // [N_X, N_X]  row-major == B-layout
    const bf16*  __restrict__ WinBf,   // [N_X, N_U]
    const bf16*  __restrict__ WoutBf,  // [N_Y, N_X]
    float*       __restrict__ y_out,   // [N_B, N_Y] for step t-1
    int has_y)
{
  __shared__ bf16 u_lds[N_B * N_U];
  const int tid = threadIdx.x;
  for (int i = tid; i < N_B * N_U; i += 256) u_lds[i] = (bf16)u[i];
  __syncthreads();

  const int wave = tid >> 5;
  const int lane = tid & 31;
  const int ln   = lane & 15;
  const int hi   = lane >> 4;

  if (wave < 4) {
    const int mt = wave;            // M-tile (rows 16*mt .. +15 of x)
    const int nc = blockIdx.x * 16; // N columns owned by this WG
    FragAB a, b;
    v8f acc = {};

    const bf16* arow = xprev + (mt * 16 + ln) * N_X;
    const bf16* brow = Wbf   + (nc + ln) * N_X;
#pragma unroll 4
    for (int kt = 0; kt < N_X / 32; ++kt) {
      const int kA = kt * 32 + (hi ? 8 : 0);
      const int kB = kt * 32 + (hi ? 8 : 0);
      a.q[0] = *(const uint4*)(arow + kA);
      a.q[1] = *(const uint4*)(arow + kA + 16);
      b.q[0] = *(const uint4*)(brow + kB);
      b.q[1] = *(const uint4*)(brow + kB + 16);
      acc = wmma_bf16(a, b, acc);
    }
    // input projection u @ W_in^T  (K = 64)
    const bf16* aurow = u_lds + (mt * 16 + ln) * N_U;
    const bf16* birow = WinBf + (nc + ln) * N_U;
#pragma unroll
    for (int kt = 0; kt < N_U / 32; ++kt) {
      const int kA = kt * 32 + (hi ? 8 : 0);
      const int kB = kt * 32 + (hi ? 8 : 0);
      a.q[0] = *(const uint4*)(aurow + kA);
      a.q[1] = *(const uint4*)(aurow + kA + 16);
      b.q[0] = *(const uint4*)(birow + kB);
      b.q[1] = *(const uint4*)(birow + kB + 16);
      acc = wmma_bf16(a, b, acc);
    }

    // epilogue: leaky tanh, write x (row-major) + x^T (history, packed b128)
    Pack8 xpk;
    const int col = nc + ln;
#pragma unroll
    for (int v = 0; v < 8; ++v) {
      const int brw = mt * 16 + hi * 8 + v;          // batch row
      const float xp = (float)xprev[brw * N_X + col];
      const float xv = (1.0f - ALPHA) * xp + ALPHA * tanhf(acc[v]);
      const bf16 xb = (bf16)xv;
      xnew[brw * N_X + col] = xb;
      xpk.h[v] = xb;
    }
    *(uint4*)(xT_t + col * N_B + mt * 16 + hi * 8) = xpk.q;
  } else if (wave == 4 && has_y && blockIdx.x < 16) {
    // y_{t-1} = x_{t-1} @ W_out^T : one 16x16 tile per WG, K = N_X
    const int mt = blockIdx.x >> 2, nt = blockIdx.x & 3;
    FragAB a, b;
    v8f acc = {};
    const bf16* arow = xprev  + (mt * 16 + ln) * N_X;
    const bf16* brow = WoutBf + (nt * 16 + ln) * N_X;
#pragma unroll 4
    for (int kt = 0; kt < N_X / 32; ++kt) {
      const int kA = kt * 32 + (hi ? 8 : 0);
      const int kB = kt * 32 + (hi ? 8 : 0);
      a.q[0] = *(const uint4*)(arow + kA);
      a.q[1] = *(const uint4*)(arow + kA + 16);
      b.q[0] = *(const uint4*)(brow + kB);
      b.q[1] = *(const uint4*)(brow + kB + 16);
      acc = wmma_bf16(a, b, acc);
    }
#pragma unroll
    for (int v = 0; v < 8; ++v)
      y_out[(mt * 16 + hi * 8 + v) * N_Y + nt * 16 + ln] = acc[v];
  }
}

// ---------------------------------------------------------------------------
// Final: y_{T-1} (wave 0 of WGs 0..15) + x_n fp32 conversion.
// ---------------------------------------------------------------------------
__global__ __launch_bounds__(256) void esn_final(
    const bf16* __restrict__ xlast, const bf16* __restrict__ WoutBf,
    float* __restrict__ y_out, float* __restrict__ xn_out)
{
  const int tid = threadIdx.x;
  const int wave = tid >> 5, lane = tid & 31;
  const int ln = lane & 15, hi = lane >> 4;
  if (wave == 0) {
    const int mt = blockIdx.x >> 2, nt = blockIdx.x & 3;
    FragAB a, b;
    v8f acc = {};
    const bf16* arow = xlast  + (mt * 16 + ln) * N_X;
    const bf16* brow = WoutBf + (nt * 16 + ln) * N_X;
#pragma unroll 4
    for (int kt = 0; kt < N_X / 32; ++kt) {
      const int kA = kt * 32 + (hi ? 8 : 0);
      const int kB = kt * 32 + (hi ? 8 : 0);
      a.q[0] = *(const uint4*)(arow + kA);
      a.q[1] = *(const uint4*)(arow + kA + 16);
      b.q[0] = *(const uint4*)(brow + kB);
      b.q[1] = *(const uint4*)(brow + kB + 16);
      acc = wmma_bf16(a, b, acc);
    }
#pragma unroll
    for (int v = 0; v < 8; ++v)
      y_out[(mt * 16 + hi * 8 + v) * N_Y + nt * 16 + ln] = acc[v];
  }
  const int gid = blockIdx.x * 256 + tid;
  for (int i = gid; i < N_B * N_X; i += 16 * 256) xn_out[i] = (float)xlast[i];
}

// ---------------------------------------------------------------------------
// X_XT = sum_{t>=trans_len} x_t^T x_t : 64x64 tiles of 16x16, K = 64*(T-t0).
// Both A and B fragments come from the N-major x^T history. One tile/wave.
// ---------------------------------------------------------------------------
__global__ __launch_bounds__(256) void esn_xxt(
    const bf16* __restrict__ xT, const int* __restrict__ trans_len,
    float* __restrict__ XXT)
{
  const int wave = threadIdx.x >> 5, lane = threadIdx.x & 31;
  const int ln = lane & 15, hi = lane >> 4;
  const int tile = blockIdx.x * 8 + wave;   // 0..4095
  const int rt = tile >> 6, ct = tile & 63;
  int t0 = *trans_len; if (t0 < 0) t0 = 0;
  FragAB a, b;
  v8f acc = {};
  for (int t = t0; t < T_STEPS; ++t) {
    const bf16* base = xT + (size_t)t * (N_X * N_B);
    const bf16* ar = base + (rt * 16 + ln) * N_B;
    const bf16* br = base + (ct * 16 + ln) * N_B;
#pragma unroll
    for (int bt = 0; bt < 2; ++bt) {
      const int kA = bt * 32 + hi * 8;
      const int kB = bt * 32 + hi * 8;
      a.q[0] = *(const uint4*)(ar + kA);
      a.q[1] = *(const uint4*)(ar + kA + 16);
      b.q[0] = *(const uint4*)(br + kB);
      b.q[1] = *(const uint4*)(br + kB + 16);
      acc = wmma_bf16(a, b, acc);
    }
  }
#pragma unroll
  for (int v = 0; v < 8; ++v)
    XXT[(rt * 16 + hi * 8 + v) * N_X + ct * 16 + ln] = acc[v];
}

// ---------------------------------------------------------------------------
// D_XT = sum_{t>=trans_len} d_t^T x_t : 4x64 tiles, A from d^T history.
// ---------------------------------------------------------------------------
__global__ __launch_bounds__(256) void esn_dxt(
    const bf16* __restrict__ xT, const bf16* __restrict__ dT,
    const int* __restrict__ trans_len, float* __restrict__ DXT)
{
  const int wave = threadIdx.x >> 5, lane = threadIdx.x & 31;
  const int ln = lane & 15, hi = lane >> 4;
  const int tile = blockIdx.x * 8 + wave;   // 0..255
  const int yt = tile >> 6, ct = tile & 63;
  int t0 = *trans_len; if (t0 < 0) t0 = 0;
  FragAB a, b;
  v8f acc = {};
  for (int t = t0; t < T_STEPS; ++t) {
    const bf16* ar = dT + (size_t)t * (N_Y * N_B) + (yt * 16 + ln) * N_B;
    const bf16* br = xT + (size_t)t * (N_X * N_B) + (ct * 16 + ln) * N_B;
#pragma unroll
    for (int bt = 0; bt < 2; ++bt) {
      const int kA = bt * 32 + hi * 8;
      const int kB = bt * 32 + hi * 8;
      a.q[0] = *(const uint4*)(ar + kA);
      a.q[1] = *(const uint4*)(ar + kA + 16);
      b.q[0] = *(const uint4*)(br + kB);
      b.q[1] = *(const uint4*)(br + kB + 16);
      acc = wmma_bf16(a, b, acc);
    }
  }
#pragma unroll
  for (int v = 0; v < 8; ++v)
    DXT[(yt * 16 + hi * 8 + v) * N_X + ct * 16 + ln] = acc[v];
}

// ---------------------------------------------------------------------------
extern "C" void kernel_launch(void* const* d_in, const int* in_sizes, int n_in,
                              void* d_out, int out_size, void* d_ws, size_t ws_size,
                              hipStream_t stream) {
  (void)in_sizes; (void)n_in; (void)out_size; (void)ws_size;
  const float* input    = (const float*)d_in[0]; // [T, N_B, N_U]
  const float* target   = (const float*)d_in[1]; // [T, N_B, N_Y]
  const float* W_in     = (const float*)d_in[2]; // [N_X, N_U]
  const float* W        = (const float*)d_in[3]; // [N_X, N_X]
  const float* W_out    = (const float*)d_in[4]; // [N_Y, N_X]
  const int*   trans_len = (const int*)d_in[5];

  float* out = (float*)d_out;
  float* Y   = out;                               // [T, N_B, N_Y]
  float* xn  = Y  + (size_t)T_STEPS * N_B * N_Y;  // [1, N_B, N_X]
  float* DXT = xn + (size_t)N_B * N_X;            // [N_Y, N_X]
  float* XXT = DXT + (size_t)N_Y * N_X;           // [N_X, N_X]

  // workspace layout (~71 MB)
  char* ws = (char*)d_ws;
  bf16* Wbf    = (bf16*)(ws);                         // 2 MB
  bf16* WinBf  = (bf16*)(ws + 2097152);               // 128 KB
  bf16* WoutBf = (bf16*)(ws + 2228224);               // 128 KB
  bf16* xbuf   = (bf16*)(ws + 2359296);               // 2 x 128 KB (double buffer)
  bf16* xT     = (bf16*)(ws + 2621440);               // 64 MB history [T][N_X][N_B]
  bf16* dT     = (bf16*)(ws + 2621440 + 67108864);    // 4 MB  [T][N_Y][N_B]

  esn_prep<<<512, 256, 0, stream>>>(W, W_in, W_out, target,
                                    Wbf, WinBf, WoutBf,
                                    xbuf + ((size_t)N_B * N_X), dT); // zero buf[1]

  for (int t = 0; t < T_STEPS; ++t) {
    const bf16* xprev = xbuf + (size_t)((t + 1) & 1) * N_B * N_X;
    bf16*       xnew  = xbuf + (size_t)(t & 1) * N_B * N_X;
    float* y_prev = Y + (size_t)(t > 0 ? t - 1 : 0) * N_B * N_Y;
    esn_step<<<64, 256, 0, stream>>>(input + (size_t)t * N_B * N_U,
                                     xprev, xnew,
                                     xT + (size_t)t * N_X * N_B,
                                     Wbf, WinBf, WoutBf,
                                     y_prev, t > 0 ? 1 : 0);
  }

  esn_final<<<16, 256, 0, stream>>>(xbuf + (size_t)((T_STEPS - 1) & 1) * N_B * N_X * 0
                                        + (size_t)N_B * N_X, // parity (T-1)&1 == 1
                                    WoutBf,
                                    Y + (size_t)(T_STEPS - 1) * N_B * N_Y, xn);
  esn_xxt<<<512, 256, 0, stream>>>(xT, trans_len, XXT);
  esn_dxt<<<32, 256, 0, stream>>>(xT, dT, trans_len, DXT);
}
